// MicroLlama_8280696946969
// MI455X (gfx1250) — matrix operand order repaired and checked by reading, not compiled, not run on hardware
//
#include <hip/hip_runtime.h>

#define SEQ     2048
#define DIM     768
#define NLAYERS 4
#define NHEADS  12
#define HD      64
#define HIDDEN  2042
#define VOCAB   32000

typedef unsigned short u16;
typedef unsigned int   u32;
typedef unsigned long long u64;
typedef __attribute__((ext_vector_type(16))) __bf16 bf16x16;
typedef __attribute__((ext_vector_type(8)))  float  floatx8;
typedef __attribute__((ext_vector_type(4)))  unsigned int u32x4;
typedef __attribute__((ext_vector_type(8)))  int          i32x8;
typedef __attribute__((ext_vector_type(4)))  int          i32x4;

#if defined(__has_builtin)
#if __has_builtin(__builtin_amdgcn_tensor_load_to_lds) && \
    __has_builtin(__builtin_amdgcn_s_wait_tensorcnt)
#define USE_TDM 1
#endif
#endif
#ifndef USE_TDM
#define USE_TDM 0
#endif

// ---------- helpers ----------
__device__ __forceinline__ u16 f2bf(float f) {
    u32 u = __float_as_uint(f);
    u += 0x7FFFu + ((u >> 16) & 1u);   // round-to-nearest-even
    return (u16)(u >> 16);
}

union FragU { uint4 u[2]; bf16x16 f; };

// A-fragment (16x32 bf16, M x K): lane row = lane&15,
// 16 contiguous K at kbase = half*16 (+ K-chunk)
__device__ __forceinline__ bf16x16 load_frag_a(const u16* rowp, int kbase) {
    FragU r;
    r.u[0] = *reinterpret_cast<const uint4*>(rowp + kbase);
    r.u[1] = *reinterpret_cast<const uint4*>(rowp + kbase + 8);
    return r.f;
}
// B-fragment (32x16 bf16, K x N) from B^T row-major (N rows, K contiguous):
// lane col = lane&15, 16 contiguous K at kbase = half*16 (+ K-chunk)
__device__ __forceinline__ bf16x16 load_frag_b(const u16* rowp, int kbase) {
    FragU r;
    r.u[0] = *reinterpret_cast<const uint4*>(rowp + kbase);
    r.u[1] = *reinterpret_cast<const uint4*>(rowp + kbase + 8);
    return r.f;
}
__device__ __forceinline__ floatx8 wmma_bf16(bf16x16 a, bf16x16 b, floatx8 c) {
    return __builtin_amdgcn_wmma_f32_16x16x32_bf16(false, a, false, b,
                                                   (short)0, c, false, false);
}

#if USE_TDM
// Issue one TDM 2-D tile load: global bf16 (row stride = `stride` elems) ->
// LDS rows padded to 40 halves (64B data + 16B pad per row).
// tensor_dim* = remaining elements from tile start, so the TDM zero-fills OOB.
__device__ __forceinline__ void tdm_load_tile(u32 lds_off, const u16* gptr,
                                              int dim0_rem, int dim1_rem,
                                              int stride) {
    u64 ga = (u64)(uintptr_t)gptr;
    u32x4 g0;
    g0[0] = 1u;                                     // count=1, user mode
    g0[1] = lds_off;                                // lds_addr (bytes)
    g0[2] = (u32)ga;                                // global_addr[31:0]
    g0[3] = (u32)((ga >> 32) & 0x1FFFFFFu) | (2u << 30);  // addr[56:32] | type=2
    i32x8 g1;
    g1[0] = (int)((1u << 16)      // data_size = 2 bytes
                | (1u << 20)      // pad_enable
                | (3u << 22)      // pad_interval: 16 DWORDs (64B) per row
                | (3u << 25));    // pad_amount: 4 DWORDs (16B)
    g1[1] = (int)(((u32)dim0_rem & 0xFFFFu) << 16);                 // dim0[15:0]
    g1[2] = (int)((((u32)dim0_rem >> 16) & 0xFFFFu)
                | (((u32)dim1_rem & 0xFFFFu) << 16));               // dim0 hi | dim1 lo
    g1[3] = (int)((((u32)dim1_rem >> 16) & 0xFFFFu) | (32u << 16)); // dim1 hi | tile_dim0=32
    g1[4] = 128;                                                    // tile_dim1=128, tile_dim2=0
    g1[5] = stride;                                                 // tensor_dim0_stride[31:0]
    g1[6] = 0;
    g1[7] = 0;
    i32x4 z4 = {0, 0, 0, 0};
    i32x8 z8 = {0, 0, 0, 0, 0, 0, 0, 0};
    // 6-arg toolchain signature: (g0, g1, g2, g3, extra, cpol)
    __builtin_amdgcn_tensor_load_to_lds(g0, g1, z4, z4, z8, 0);
}
__device__ __forceinline__ u32 lds_byte_off(const void* p) {
    return (u32)(uintptr_t)p;   // generic shared address: LDS offset in [31:0]
}
#endif

// ---------- fp32 -> bf16 bulk convert ----------
__global__ void ml_f2bf(const float* __restrict__ in, u16* __restrict__ out, long n) {
    long i = (long)blockIdx.x * blockDim.x + threadIdx.x;
    if (i < n) out[i] = f2bf(in[i]);
}

// ---------- embedding gather ----------
__global__ void ml_embed(const int* __restrict__ tokens,
                         const float* __restrict__ emb,
                         float* __restrict__ h) {
    int s = blockIdx.x;
    int tok = tokens[s];
    for (int d = threadIdx.x; d < DIM; d += blockDim.x)
        h[(size_t)s * DIM + d] = emb[(size_t)tok * DIM + d];
}

// ---------- RMSNorm (fp32 in, bf16 out) ----------
__global__ void ml_rmsnorm(const float* __restrict__ x,
                           const float* __restrict__ w,
                           u16* __restrict__ out) {
    __shared__ float red[256];
    int s = blockIdx.x;
    const float* xr = x + (size_t)s * DIM;
    float ss = 0.f;
    for (int d = threadIdx.x; d < DIM; d += 256) { float v = xr[d]; ss += v * v; }
    red[threadIdx.x] = ss;
    __syncthreads();
    for (int o = 128; o > 0; o >>= 1) {
        if (threadIdx.x < o) red[threadIdx.x] += red[threadIdx.x + o];
        __syncthreads();
    }
    float scale = rsqrtf(red[0] / (float)DIM + 1e-6f);
    u16* outr = out + (size_t)s * DIM;
    for (int d = threadIdx.x; d < DIM; d += 256)
        outr[d] = f2bf(xr[d] * scale * w[d]);
}

// ---------- RoPE (in place fp32, consecutive pairs) ----------
__global__ void ml_rope(float* __restrict__ x) {
    int idx = blockIdx.x * blockDim.x + threadIdx.x;
    if (idx >= SEQ * NHEADS * (HD / 2)) return;
    int p = idx & 31;
    int h = (idx >> 5) % NHEADS;
    int s = idx / (NHEADS * 32);
    float freq = __powf(10000.f, -(2.f * (float)p) / (float)HD);
    float ang = (float)s * freq;
    float sn, cs;
    __sincosf(ang, &sn, &cs);
    float* b = x + (size_t)s * DIM + h * HD + 2 * p;
    float a0 = b[0], b0 = b[1];
    b[0] = a0 * cs - b0 * sn;
    b[1] = a0 * sn + b0 * cs;
}

// ---------- SiLU(f1) * f3 -> bf16 g ----------
__global__ void ml_silu_mul(const float* __restrict__ f1,
                            const float* __restrict__ f3,
                            u16* __restrict__ g, long n) {
    long i = (long)blockIdx.x * blockDim.x + threadIdx.x;
    if (i < n) {
        float a = f1[i];
        float s = a / (1.f + __expf(-a));
        g[i] = f2bf(s * f3[i]);
    }
}

// ---------- WMMA GEMM: C[M,N](f32) = (Res? Res:0) + Abf16[M,K] * Bbf16[N,K]^T ----
#define BM 128
#define BN 128
#define BK 32
#define LDSA 40   // 32 data + 8 pad halves (= 64B + 16B), matches TDM padding

#if !USE_TDM
__device__ __forceinline__ void stage_tiles_bf16(const u16* __restrict__ A,
                                                 const u16* __restrict__ B,
                                                 u16* As, u16* Bs,
                                                 int m0, int n0, int k0,
                                                 int M, int N, int K, int tid) {
    int rbase = tid >> 4;
    int col2  = (tid & 15) * 2;   // K always even
#pragma unroll
    for (int p = 0; p < 8; p++) {
        int r = p * 16 + rbase;
        u32 av = 0, bv = 0;
        int gm = m0 + r, gn = n0 + r;
        if (gm < M && (k0 + col2) < K)
            av = *reinterpret_cast<const u32*>(A + (size_t)gm * K + k0 + col2);
        if (gn < N && (k0 + col2) < K)
            bv = *reinterpret_cast<const u32*>(B + (size_t)gn * K + k0 + col2);
        *reinterpret_cast<u32*>(&As[r * LDSA + col2]) = av;
        *reinterpret_cast<u32*>(&Bs[r * LDSA + col2]) = bv;
    }
}
#endif

__global__ __launch_bounds__(256) void ml_gemm(const u16* __restrict__ A,
                                               const u16* __restrict__ B,
                                               const float* Res,
                                               float* C,
                                               int M, int N, int K) {
    __shared__ u16 As[2][BM * LDSA];
    __shared__ u16 Bs[2][BN * LDSA];

    const int tid   = threadIdx.x;
    const int lane  = tid & 31;
    const int wave  = tid >> 5;
    const int halfw = lane >> 4;
    const int lo    = lane & 15;
    const int wm    = wave & 1;     // 2 waves over M (64 each)
    const int wn    = wave >> 1;    // 4 waves over N (32 each)
    const int m0    = blockIdx.y * BM;
    const int n0    = blockIdx.x * BN;

    floatx8 acc[4][2];
#pragma unroll
    for (int i = 0; i < 4; i++)
#pragma unroll
        for (int j = 0; j < 2; j++)
#pragma unroll
            for (int e = 0; e < 8; e++) acc[i][j][e] = 0.f;

    const int nStages = (K + BK - 1) / BK;

#if USE_TDM
    if (wave == 0) {
        tdm_load_tile(lds_byte_off(&As[0][0]), A + (size_t)m0 * K, K, M - m0, K);
        tdm_load_tile(lds_byte_off(&Bs[0][0]), B + (size_t)n0 * K, K, N - n0, K);
        __builtin_amdgcn_s_wait_tensorcnt(0);
    }
#else
    stage_tiles_bf16(A, B, As[0], Bs[0], m0, n0, 0, M, N, K, tid);
#endif
    __syncthreads();

    for (int ks = 0; ks < nStages; ks++) {
        const int buf = ks & 1;
        if (ks + 1 < nStages) {
            const int k1 = (ks + 1) * BK;
#if USE_TDM
            if (wave == 0) {
                tdm_load_tile(lds_byte_off(&As[buf ^ 1][0]),
                              A + (size_t)m0 * K + k1, K - k1, M - m0, K);
                tdm_load_tile(lds_byte_off(&Bs[buf ^ 1][0]),
                              B + (size_t)n0 * K + k1, K - k1, N - n0, K);
            }
#else
            stage_tiles_bf16(A, B, As[buf ^ 1], Bs[buf ^ 1], m0, n0, k1,
                             M, N, K, tid);
#endif
        }

        const u16* as = As[buf];
        const u16* bs = Bs[buf];
        bf16x16 af[4];
        bf16x16 bfrag[2];
#pragma unroll
        for (int i = 0; i < 4; i++)
            af[i] = load_frag_a(as + (wm * 64 + i * 16 + lo) * LDSA, halfw * 16);
#pragma unroll
        for (int j = 0; j < 2; j++)
            bfrag[j] = load_frag_b(bs + (wn * 32 + j * 16 + lo) * LDSA, halfw * 16);
#pragma unroll
        for (int i = 0; i < 4; i++)
#pragma unroll
            for (int j = 0; j < 2; j++)
                acc[i][j] = wmma_bf16(af[i], bfrag[j], acc[i][j]);

#if USE_TDM
        if (wave == 0) __builtin_amdgcn_s_wait_tensorcnt(0);
#endif
        __syncthreads();
    }

    // epilogue: C layout — VGPR e -> row e + half*8, col = lane&15
#pragma unroll
    for (int i = 0; i < 4; i++)
#pragma unroll
        for (int j = 0; j < 2; j++)
#pragma unroll
            for (int e = 0; e < 8; e++) {
                int gm = m0 + wm * 64 + i * 16 + e + halfw * 8;
                int gn = n0 + wn * 32 + j * 16 + lo;
                if (gm < M && gn < N) {
                    size_t off = (size_t)gm * N + gn;
                    float r = Res ? Res[off] : 0.f;
                    C[off] = r + acc[i][j][e];
                }
            }
}

// ---------- flash attention (causal), WMMA for QK^T and PV ----------
#define ATW 4
#define ATT_WAVE_HALVES 6656   // Qf 16*72 + Kf 32*72 + Pf 16*40 + Vt 64*40

__global__ __launch_bounds__(128) void ml_attn(const float* __restrict__ q,
                                               const float* __restrict__ k,
                                               const float* __restrict__ v,
                                               u16* __restrict__ o) {
    __shared__ u16 smem[ATW * ATT_WAVE_HALVES];
    const int wave  = threadIdx.x >> 5;
    const int lane  = threadIdx.x & 31;
    const int halfw = lane >> 4;
    const int lo    = lane & 15;
    const int head  = blockIdx.y;
    const int q0    = blockIdx.x * (ATW * 16) + wave * 16;

    u16* Qf = smem + wave * ATT_WAVE_HALVES;  // [16][72]
    u16* Kf = Qf + 16 * 72;                   // [32][72]
    u16* Pf = Kf + 32 * 72;                   // [16][40]
    u16* Vt = Pf + 16 * 40;                   // [64][40]  (V transposed: [d][key])

    const size_t hoff = (size_t)head * HD;

    for (int idx = lane; idx < 16 * HD; idx += 32) {
        int r = idx >> 6, d = idx & 63;
        Qf[r * 72 + d] = f2bf(q[(size_t)(q0 + r) * DIM + hoff + d]);
    }

    float mrow[8], lrow[8];
    floatx8 oacc[4];
#pragma unroll
    for (int j = 0; j < 8; j++) { mrow[j] = -1e30f; lrow[j] = 0.f; }
#pragma unroll
    for (int t = 0; t < 4; t++)
#pragma unroll
        for (int j = 0; j < 8; j++) oacc[t][j] = 0.f;

    const float scale = 0.125f;  // 1/sqrt(64)
    const int nChunks = (q0 + 16 + 31) >> 5;

    for (int c = 0; c < nChunks; c++) {
        const int kstart = c * 32;
        for (int idx = lane; idx < 32 * HD; idx += 32) {
            int t = idx >> 6, d = idx & 63;
            int kr = kstart + t;
            Kf[t * 72 + d] = f2bf(k[(size_t)kr * DIM + hoff + d]);
            Vt[d * 40 + t] = f2bf(v[(size_t)kr * DIM + hoff + d]);
        }
        asm volatile("s_wait_dscnt 0" ::: "memory");  // same-wave DS store->load

        floatx8 s0, s1;
#pragma unroll
        for (int e = 0; e < 8; e++) { s0[e] = 0.f; s1[e] = 0.f; }
#pragma unroll
        for (int kk = 0; kk < 2; kk++) {
            bf16x16 aq = load_frag_a(Qf + lo * 72, kk * 32 + halfw * 16);
            bf16x16 b0 = load_frag_b(Kf + lo * 72, kk * 32 + halfw * 16);
            bf16x16 b1 = load_frag_b(Kf + (16 + lo) * 72, kk * 32 + halfw * 16);
            s0 = wmma_bf16(aq, b0, s0);
            s1 = wmma_bf16(aq, b1, s1);
        }

#pragma unroll
        for (int j = 0; j < 8; j++) {
            int qi = q0 + j + halfw * 8;
            float e0 = (kstart + lo      <= qi) ? s0[j] * scale : -1e30f;
            float e1 = (kstart + 16 + lo <= qi) ? s1[j] * scale : -1e30f;
            float rm = fmaxf(e0, e1);
#pragma unroll
            for (int off = 8; off >= 1; off >>= 1)
                rm = fmaxf(rm, __shfl_xor(rm, off, 32));
            float mnew  = fmaxf(mrow[j], rm);
            float alpha = __expf(mrow[j] - mnew);
            float p0 = __expf(e0 - mnew);
            float p1 = __expf(e1 - mnew);
            float rs = p0 + p1;
#pragma unroll
            for (int off = 8; off >= 1; off >>= 1)
                rs += __shfl_xor(rs, off, 32);
            lrow[j] = lrow[j] * alpha + rs;
            mrow[j] = mnew;
#pragma unroll
            for (int t = 0; t < 4; t++) oacc[t][j] *= alpha;
            int prow = j + halfw * 8;
            Pf[prow * 40 + lo]      = f2bf(p0);
            Pf[prow * 40 + 16 + lo] = f2bf(p1);
        }
        asm volatile("s_wait_dscnt 0" ::: "memory");

        bf16x16 ap = load_frag_a(Pf + lo * 40, halfw * 16);
#pragma unroll
        for (int t = 0; t < 4; t++) {
            bf16x16 bv = load_frag_b(Vt + (t * 16 + lo) * 40, halfw * 16);
            oacc[t] = wmma_bf16(ap, bv, oacc[t]);
        }
    }

#pragma unroll
    for (int t = 0; t < 4; t++)
#pragma unroll
        for (int j = 0; j < 8; j++) {
            int qi = q0 + j + halfw * 8;
            o[(size_t)qi * DIM + hoff + t * 16 + lo] = f2bf(oacc[t][j] / lrow[j]);
        }
}

// ---------- host orchestration ----------
extern "C" void kernel_launch(void* const* d_in, const int* in_sizes, int n_in,
                              void* d_out, int out_size, void* d_ws, size_t ws_size,
                              hipStream_t stream) {
    (void)in_sizes; (void)n_in; (void)out_size; (void)ws_size;
    const int*   tokens = (const int*)d_in[0];
    const float* emb  = (const float*)d_in[1];
    const float* wq   = (const float*)d_in[2];
    const float* wk   = (const float*)d_in[3];
    const float* wv   = (const float*)d_in[4];
    const float* wo   = (const float*)d_in[5];
    const float* w1   = (const float*)d_in[6];
    const float* w2   = (const float*)d_in[7];
    const float* w3   = (const float*)d_in[8];
    const float* anw  = (const float*)d_in[9];
    const float* fnw  = (const float*)d_in[10];
    const float* nw   = (const float*)d_in[11];
    const float* outw = (const float*)d_in[12];
    float* logits = (float*)d_out;

    const size_t SD = (size_t)SEQ * DIM;
    const size_t SH = (size_t)SEQ * HIDDEN;
    const size_t WQKVO = (size_t)NLAYERS * DIM * DIM;      // per q/k/v/o tensor
    const size_t WFF   = (size_t)NLAYERS * HIDDEN * DIM;   // per w1/w2/w3 tensor
    const size_t WOUT  = (size_t)VOCAB * DIM;

    char* p = (char*)d_ws;
    auto alloc = [&](size_t bytes) {
        void* r = (void*)p;
        p += (bytes + 255) & ~(size_t)255;
        return r;
    };
    float* h  = (float*)alloc(SD * 4);
    float* qb = (float*)alloc(SD * 4);
    float* kb = (float*)alloc(SD * 4);
    float* vb = (float*)alloc(SD * 4);
    float* f1 = (float*)alloc(SH * 4);
    float* f3 = (float*)alloc(SH * 4);
    u16* xn  = (u16*)alloc(SD * 2);
    u16* ao  = (u16*)alloc(SD * 2);
    u16* g   = (u16*)alloc(SH * 2);
    u16* bwq = (u16*)alloc(WQKVO * 2);
    u16* bwk = (u16*)alloc(WQKVO * 2);
    u16* bwv = (u16*)alloc(WQKVO * 2);
    u16* bwo = (u16*)alloc(WQKVO * 2);
    u16* bw1 = (u16*)alloc(WFF * 2);
    u16* bw2 = (u16*)alloc(WFF * 2);
    u16* bw3 = (u16*)alloc(WFF * 2);
    u16* bow = (u16*)alloc(WOUT * 2);

    auto cvt = [&](const float* src, u16* dst, size_t n) {
        ml_f2bf<<<(int)((n + 255) / 256), 256, 0, stream>>>(src, dst, (long)n);
    };
    cvt(wq, bwq, WQKVO);  cvt(wk, bwk, WQKVO);
    cvt(wv, bwv, WQKVO);  cvt(wo, bwo, WQKVO);
    cvt(w1, bw1, WFF);    cvt(w2, bw2, WFF);    cvt(w3, bw3, WFF);
    cvt(outw, bow, WOUT);

    const dim3 blk(256);
    const dim3 g768((DIM + BN - 1) / BN, (SEQ + BM - 1) / BM);
    const dim3 ghid((HIDDEN + BN - 1) / BN, (SEQ + BM - 1) / BM);
    const dim3 gvoc((VOCAB + BN - 1) / BN, (SEQ + BM - 1) / BM);
    const int ropeN = SEQ * NHEADS * (HD / 2);
    const long nHid = (long)SEQ * HIDDEN;

    ml_embed<<<SEQ, 256, 0, stream>>>(tokens, emb, h);

    for (int L = 0; L < NLAYERS; L++) {
        const u16* wq_l = bwq + (size_t)L * DIM * DIM;
        const u16* wk_l = bwk + (size_t)L * DIM * DIM;
        const u16* wv_l = bwv + (size_t)L * DIM * DIM;
        const u16* wo_l = bwo + (size_t)L * DIM * DIM;
        const u16* w1_l = bw1 + (size_t)L * HIDDEN * DIM;
        const u16* w2_l = bw2 + (size_t)L * DIM * HIDDEN;
        const u16* w3_l = bw3 + (size_t)L * HIDDEN * DIM;

        ml_rmsnorm<<<SEQ, 256, 0, stream>>>(h, anw + L * DIM, xn);
        ml_gemm<<<g768, blk, 0, stream>>>(xn, wq_l, nullptr, qb, SEQ, DIM, DIM);
        ml_gemm<<<g768, blk, 0, stream>>>(xn, wk_l, nullptr, kb, SEQ, DIM, DIM);
        ml_gemm<<<g768, blk, 0, stream>>>(xn, wv_l, nullptr, vb, SEQ, DIM, DIM);
        ml_rope<<<(ropeN + 255) / 256, 256, 0, stream>>>(qb);
        ml_rope<<<(ropeN + 255) / 256, 256, 0, stream>>>(kb);
        ml_attn<<<dim3(SEQ / (ATW * 16), NHEADS), 128, 0, stream>>>(qb, kb, vb, ao);
        ml_gemm<<<g768, blk, 0, stream>>>(ao, wo_l, h, h, SEQ, DIM, DIM);
        ml_rmsnorm<<<SEQ, 256, 0, stream>>>(h, fnw + L * DIM, xn);
        ml_gemm<<<ghid, blk, 0, stream>>>(xn, w1_l, nullptr, f1, SEQ, HIDDEN, DIM);
        ml_gemm<<<ghid, blk, 0, stream>>>(xn, w3_l, nullptr, f3, SEQ, HIDDEN, DIM);
        ml_silu_mul<<<(int)((nHid + 255) / 256), 256, 0, stream>>>(f1, f3, g, nHid);
        ml_gemm<<<g768, blk, 0, stream>>>(g, w2_l, h, h, SEQ, DIM, HIDDEN);
    }

    ml_rmsnorm<<<SEQ, 256, 0, stream>>>(h, nw, xn);
    ml_gemm<<<gvoc, blk, 0, stream>>>(xn, bow, nullptr, logits, SEQ, VOCAB, DIM);
}
